// VisionEncoder_34943853920521
// MI455X (gfx1250) — compile-verified, matrix operand order checked by reading, not run on hardware
//
#include <hip/hip_runtime.h>
#include <hip/hip_bf16.h>
#include <math.h>

// ---------------- model constants ----------------
#define DVEC   1152      // vision dim
#define NTOK   4096      // tokens
#define TDIM   3456      // 3*D
#define MLPH   4304      // mlp hidden
#define MLPP   4352      // mlp hidden padded to mult of 64
#define KPATCH 608       // 3*14*14=588 padded to mult of 32
#define KREAL  588
#define IMGW   896
#define PATCH  14
#define LBLK   2

typedef __bf16 bf16_t;
typedef __attribute__((ext_vector_type(16))) __bf16 v16bf;
typedef __attribute__((ext_vector_type(8)))  __bf16 v8bf;
typedef __attribute__((ext_vector_type(8)))  float  v8f;
typedef __attribute__((ext_vector_type(4)))  unsigned int v4u;
typedef __attribute__((ext_vector_type(8)))  int v8i;
typedef __attribute__((ext_vector_type(4)))  int v4i;

#if __has_builtin(__builtin_amdgcn_tensor_load_to_lds) && \
    __has_builtin(__builtin_amdgcn_s_wait_tensorcnt)
#define USE_TDM 1
#else
#define USE_TDM 0
#endif

// ---------------- helpers ----------------
__device__ __forceinline__ bf16_t f2bf(float f) {
    unsigned u = __float_as_uint(f);
    unsigned r = (u + 0x7FFFu + ((u >> 16) & 1u)) >> 16;   // round-to-nearest-even
    unsigned short s = (unsigned short)r;
    return __builtin_bit_cast(bf16_t, s);
}

__device__ __forceinline__ v16bf cat16(v8bf lo, v8bf hi) {
    return __builtin_shufflevector(lo, hi, 0,1,2,3,4,5,6,7,8,9,10,11,12,13,14,15);
}

__device__ __forceinline__ float gelu_exact(float v) {
    return 0.5f * v * (1.0f + erff(v * 0.70710678118654752f));
}

__device__ __forceinline__ float block_reduce_sum(float v, float* red) {
    int tid = threadIdx.x;
    red[tid] = v; __syncthreads();
    for (int s = 128; s > 0; s >>= 1) {
        if (tid < s) red[tid] += red[tid + s];
        __syncthreads();
    }
    float r = red[0]; __syncthreads();
    return r;
}

__device__ __forceinline__ float block_reduce_max(float v, float* red) {
    int tid = threadIdx.x;
    red[tid] = v; __syncthreads();
    for (int s = 128; s > 0; s >>= 1) {
        if (tid < s) red[tid] = fmaxf(red[tid], red[tid + s]);
        __syncthreads();
    }
    float r = red[0]; __syncthreads();
    return r;
}

#if USE_TDM
// ---------------- Tensor Data Mover: DMA a [rows x 32] bf16 tile into LDS ----------------
// D# per CDNA5 ISA:
//  group0: count=1 | lds_addr | global_addr(57b) | type=2
//  group1: data_size=2B, tensor_dim0=32, tensor_dim1=rows, tile_dim0=32,
//          tile_dim1=rows, tensor_dim0_stride=ld (elements); no pad/iterate/multicast
__device__ __forceinline__ void tdm_load_tile(unsigned lds_byte_addr,
                                              const bf16_t* gptr,
                                              unsigned rows, unsigned ld_elems)
{
    unsigned long long ga = (unsigned long long)(uintptr_t)gptr;
    const unsigned td0 = 32u;   // tensor/tile dim0 (K elements)
    v4u g0 = { 1u,                                   // count=1, no gather
               lds_byte_addr,
               (unsigned)(ga & 0xFFFFFFFFu),
               (unsigned)((ga >> 32) & 0x01FFFFFFu) | (2u << 30) }; // type=2
    v8i g1 = { (int)(1u << 16),                      // data_size=1 -> 2 bytes
               (int)((td0 & 0xFFFFu) << 16),         // tensor_dim0[15:0]
               (int)((td0 >> 16) | (rows << 16)),    // tensor_dim0[31:16] | tensor_dim1[15:0]
               (int)((rows >> 16) | (td0 << 16)),    // tensor_dim1[31:16] | tile_dim0
               (int)rows,                            // tile_dim1 (tile_dim2=0)
               (int)ld_elems,                        // tensor_dim0_stride[31:0]
               0, 0 };
    v4i gz = { 0, 0, 0, 0 };
#if defined(__clang_major__) && (__clang_major__ >= 23)
    v8i gz8 = { 0, 0, 0, 0, 0, 0, 0, 0 };
    __builtin_amdgcn_tensor_load_to_lds(g0, g1, gz, gz, gz8, 0);
#else
    __builtin_amdgcn_tensor_load_to_lds(g0, g1, gz, gz, 0);
#endif
}
#endif

// ---------------- WMMA GEMM (TN form) ----------------
// C[M,N] = A[M,K](bf16, row-major) x B[N,K](bf16, row-major)^T
//          (+bias[N]) (+residual[M,N]) (gelu?)
// Tile: 128(M) x 64(N) per 128-thread block (4 waves), K stepped by 32.
// Wave w owns rows [w*32, w*32+32): 2 A-fragments x 4 B-fragments = 8 WMMA/K-step.
// Tiles staged into LDS by the Tensor Data Mover (TDM) when available.
// M mult of 128; N mult of 64; K mult of 32; lda/ldb mult of 8.
__global__ __launch_bounds__(128)
void gemm_bf16_wmma(const bf16_t* __restrict__ A, int lda,
                    const bf16_t* __restrict__ B, int ldb,
                    const float*  __restrict__ bias,
                    const float*  __restrict__ residual,
                    float*        __restrict__ Cf,
                    bf16_t*       __restrict__ Cb,
                    int ldc, int M, int N, int K, int gelu_flag)
{
    __shared__ __align__(16) bf16_t As[128][32];
    __shared__ __align__(16) bf16_t Bs[64][32];

    const int tid  = threadIdx.x;
    const int wave = tid >> 5;
    const int lane = tid & 31;
    const int r16  = lane & 15;
    const int hi   = lane >> 4;      // selects K half per CDNA5 16-bit layout
    const int m0   = blockIdx.y * 128;
    const int n0   = blockIdx.x * 64;

#if USE_TDM
    const unsigned ldsA = (unsigned)(uintptr_t)(&As[0][0]); // low 32 bits = LDS offset
    const unsigned ldsB = (unsigned)(uintptr_t)(&Bs[0][0]);
#endif

    v8f acc0[4] = {};
    v8f acc1[4] = {};

    for (int k0 = 0; k0 < K; k0 += 32) {
#if USE_TDM
        // Tensor Data Mover stages both tiles; wave 0 issues + waits, all sync.
        if (wave == 0) {
            tdm_load_tile(ldsA, A + (size_t)m0 * lda + k0, 128u, (unsigned)lda);
            tdm_load_tile(ldsB, B + (size_t)n0 * ldb + k0,  64u, (unsigned)ldb);
            __builtin_amdgcn_s_wait_tensorcnt(0);
        }
#else
        // fallback: stage through VGPRs
        {
            const uint4* src = (const uint4*)(A + (size_t)(m0 + tid) * lda + k0);
            uint4* dst = (uint4*)(&As[tid][0]);
            dst[0] = src[0]; dst[1] = src[1]; dst[2] = src[2]; dst[3] = src[3];
        }
        {
            int row = tid >> 1;
            int seg = (tid & 1) * 16;
            const uint4* src = (const uint4*)(B + (size_t)(n0 + row) * ldb + k0 + seg);
            uint4* dst = (uint4*)(&Bs[row][seg]);
            dst[0] = src[0]; dst[1] = src[1];
        }
#endif
        __syncthreads();

        // A fragments: lane's 16 elements = two contiguous 8-element K runs
        const int arow0 = wave * 32 + r16;
        v16bf a0 = cat16(*(const v8bf*)&As[arow0     ][hi * 8],
                         *(const v8bf*)&As[arow0     ][16 + hi * 8]);
        v16bf a1 = cat16(*(const v8bf*)&As[arow0 + 16][hi * 8],
                         *(const v8bf*)&As[arow0 + 16][16 + hi * 8]);

        #pragma unroll
        for (int c = 0; c < 4; ++c) {
            const int brow = c * 16 + r16;
            v16bf b = cat16(*(const v8bf*)&Bs[brow][hi * 8],
                            *(const v8bf*)&Bs[brow][16 + hi * 8]);
            acc0[c] = __builtin_amdgcn_wmma_f32_16x16x32_bf16(false, a0, false, b, (short)0, acc0[c], false, false);
            acc1[c] = __builtin_amdgcn_wmma_f32_16x16x32_bf16(false, a1, false, b, (short)0, acc1[c], false, false);
        }
        __syncthreads();
    }

    // epilogue: D layout -> VGPR r holds row (frag*16 + hi*8 + r), col = c*16 + r16
    #pragma unroll
    for (int f = 0; f < 2; ++f) {
        const int rowb = m0 + wave * 32 + f * 16 + hi * 8;
        #pragma unroll
        for (int c = 0; c < 4; ++c) {
            const int col = n0 + c * 16 + r16;
            const float bv = bias ? bias[col] : 0.0f;
            #pragma unroll
            for (int r = 0; r < 8; ++r) {
                const int row = rowb + r;
                float v = (f == 0 ? acc0[c][r] : acc1[c][r]) + bv;
                if (residual) v += residual[(size_t)row * ldc + col];
                if (gelu_flag) v = gelu_exact(v);
                if (Cf) Cf[(size_t)row * ldc + col] = v;
                if (Cb) Cb[(size_t)row * ldc + col] = f2bf(v);
            }
        }
    }
}

// ---------------- cast + zero-pad weights: W[rows,cols] f32 -> out[Rp,Cp] bf16 ----------------
__global__ void cast_pad_bf16(const float* __restrict__ W, int rows, int cols,
                              bf16_t* __restrict__ out, int Rp, int Cp)
{
    size_t i = (size_t)blockIdx.x * blockDim.x + threadIdx.x;
    size_t total = (size_t)Rp * Cp;
    if (i >= total) return;
    int r = (int)(i / Cp);
    int c = (int)(i % Cp);
    float v = (r < rows && c < cols) ? W[(size_t)r * cols + c] : 0.0f;
    out[i] = f2bf(v);
}

__global__ void pad_bias_kernel(const float* __restrict__ b, float* __restrict__ o,
                                int n, int npad)
{
    int i = blockIdx.x * blockDim.x + threadIdx.x;
    if (i >= npad) return;
    o[i] = (i < n) ? b[i] : 0.0f;
}

// ---------------- im2col (patchify) ----------------
__global__ void im2col_bf16(const float* __restrict__ img, bf16_t* __restrict__ Xp)
{
    size_t i = (size_t)blockIdx.x * blockDim.x + threadIdx.x;
    size_t total = (size_t)NTOK * KPATCH;
    if (i >= total) return;
    int n = (int)(i / KPATCH);
    int k = (int)(i % KPATCH);
    float v = 0.0f;
    if (k < KREAL) {
        int c   = k / (PATCH * PATCH);
        int rem = k % (PATCH * PATCH);
        int py  = rem / PATCH;
        int px  = rem % PATCH;
        int prow = n >> 6;      // 64 patches per image row
        int pcol = n & 63;
        v = img[(size_t)c * IMGW * IMGW + (size_t)(prow * PATCH + py) * IMGW + (pcol * PATCH + px)];
    }
    Xp[i] = f2bf(v);
}

__global__ void add_pos_kernel(float* __restrict__ x, const float* __restrict__ pos)
{
    size_t i = (size_t)blockIdx.x * blockDim.x + threadIdx.x;
    if (i >= (size_t)NTOK * DVEC) return;
    int n = (int)(i / DVEC);
    int d = (int)(i % DVEC);
    x[i] += pos[(size_t)d * NTOK + n];
}

// ---------------- layernorm (one block per token row) ----------------
__global__ __launch_bounds__(256)
void layernorm_kernel(const float* __restrict__ x, const float* __restrict__ w,
                      const float* __restrict__ b, bf16_t* __restrict__ obf,
                      float* __restrict__ of)
{
    __shared__ float red[256];
    const int tid = threadIdx.x;
    const int row = blockIdx.x;
    const float* xr = x + (size_t)row * DVEC;

    float s = 0.0f;
    for (int i = tid; i < DVEC; i += 256) s += xr[i];
    float mean = block_reduce_sum(s, red) * (1.0f / DVEC);

    float vs = 0.0f;
    for (int i = tid; i < DVEC; i += 256) { float d = xr[i] - mean; vs += d * d; }
    float var = block_reduce_sum(vs, red) * (1.0f / DVEC);
    float rstd = rsqrtf(var + 1e-6f);

    for (int i = tid; i < DVEC; i += 256) {
        float y = (xr[i] - mean) * rstd * w[i] + b[i];
        if (obf) obf[(size_t)row * DVEC + i] = f2bf(y);
        if (of)  of[(size_t)row * DVEC + i] = y;
    }
}

// ---------------- softmax row (S f32 -> P bf16) ----------------
__global__ __launch_bounds__(256)
void softmax_kernel(const float* __restrict__ S, bf16_t* __restrict__ P)
{
    __shared__ float red[256];
    const int tid = threadIdx.x;
    const int row = blockIdx.x;
    const float* sr = S + (size_t)row * NTOK;

    float m = -1e30f;
    for (int i = tid; i < NTOK; i += 256) m = fmaxf(m, sr[i]);
    float mx = block_reduce_max(m, red);

    float s = 0.0f;
    for (int i = tid; i < NTOK; i += 256) s += __expf(sr[i] - mx);
    float inv = 1.0f / block_reduce_sum(s, red);

    for (int i = tid; i < NTOK; i += 256)
        P[(size_t)row * NTOK + i] = f2bf(__expf(sr[i] - mx) * inv);
}

// ---------------- strided extract / cast (optionally transposed) ----------------
// src[r, col0 + c] (ld) * scale -> dst  (transpose: dst[c*rows + r], else dst[r*cols + c])
__global__ void extract_cast_kernel(const float* __restrict__ src, int ld, int col0,
                                    bf16_t* __restrict__ dst, int rows, int cols,
                                    float scale, int transpose)
{
    size_t i = (size_t)blockIdx.x * blockDim.x + threadIdx.x;
    if (i >= (size_t)rows * cols) return;
    int r = (int)(i / cols);
    int c = (int)(i % cols);
    float v = src[(size_t)r * ld + col0 + c] * scale;
    if (transpose) dst[(size_t)c * rows + r] = f2bf(v);
    else           dst[i] = f2bf(v);
}

// ---------------- host orchestration ----------------
static inline int ceil_div(size_t a, int b) { return (int)((a + b - 1) / b); }

extern "C" void kernel_launch(void* const* d_in, const int* in_sizes, int n_in,
                              void* d_out, int out_size, void* d_ws, size_t ws_size,
                              hipStream_t stream)
{
    const float* img     = (const float*)d_in[0];
    const float* patch_w = (const float*)d_in[1];
    const float* patch_b = (const float*)d_in[2];
    const float* pos     = (const float*)d_in[3];
    const float* ln1_w   = (const float*)d_in[4];
    const float* ln1_b   = (const float*)d_in[5];
    const float* qkv_w   = (const float*)d_in[6];
    const float* qkv_b   = (const float*)d_in[7];
    const float* proj_w  = (const float*)d_in[8];
    const float* proj_b  = (const float*)d_in[9];
    const float* ln2_w   = (const float*)d_in[10];
    const float* ln2_b   = (const float*)d_in[11];
    const float* fc1_w   = (const float*)d_in[12];
    const float* fc1_b   = (const float*)d_in[13];
    const float* fc2_w   = (const float*)d_in[14];
    const float* fc2_b   = (const float*)d_in[15];
    const float* post_w  = (const float*)d_in[16];
    const float* post_b  = (const float*)d_in[17];
    float* out = (float*)d_out;

    // ---- workspace carve-out (256B aligned) ----
    char* p = (char*)d_ws;
    auto alloc = [&](size_t bytes) -> void* {
        void* r = (void*)p;
        p += (bytes + 255) & ~(size_t)255;
        return r;
    };
    // weights (B operand layout = [N][K] row-major = native torch [out][in])
    bf16_t* wp_bf   = (bf16_t*)alloc((size_t)DVEC * KPATCH * 2);
    bf16_t* wqkv_bf = (bf16_t*)alloc((size_t)LBLK * TDIM * DVEC * 2);
    bf16_t* wproj_bf= (bf16_t*)alloc((size_t)LBLK * DVEC * DVEC * 2);
    bf16_t* wfc1_bf = (bf16_t*)alloc((size_t)LBLK * MLPP * DVEC * 2);
    bf16_t* wfc2_bf = (bf16_t*)alloc((size_t)LBLK * DVEC * MLPP * 2);
    float*  fc1b_pad= (float*) alloc((size_t)LBLK * MLPP * 4);
    // activations
    bf16_t* xp_bf   = (bf16_t*)alloc((size_t)NTOK * KPATCH * 2);
    float*  x       = (float*) alloc((size_t)NTOK * DVEC * 4);
    bf16_t* ln_bf   = (bf16_t*)alloc((size_t)NTOK * DVEC * 2);
    float*  qkv     = (float*) alloc((size_t)NTOK * TDIM * 4);
    bf16_t* qb      = (bf16_t*)alloc((size_t)NTOK * DVEC * 2);   // Q * scale   [N,D]
    bf16_t* kb      = (bf16_t*)alloc((size_t)NTOK * DVEC * 2);   // K           [N,D]
    bf16_t* vtb     = (bf16_t*)alloc((size_t)DVEC * NTOK * 2);   // V^T         [D,N]
    float*  S       = (float*) alloc((size_t)NTOK * NTOK * 4);
    bf16_t* P       = (bf16_t*)alloc((size_t)NTOK * NTOK * 2);
    bf16_t* o_bf    = (bf16_t*)alloc((size_t)NTOK * DVEC * 2);
    bf16_t* h_bf    = (bf16_t*)alloc((size_t)NTOK * MLPP * 2);

    const int T = 256;

    // ---- weight conversion: cast + zero-pad only (no transposes; GEMM is TN) ----
    cast_pad_bf16<<<ceil_div((size_t)DVEC * KPATCH, T), T, 0, stream>>>(
        patch_w, DVEC, KREAL, wp_bf, DVEC, KPATCH);
    for (int l = 0; l < LBLK; ++l) {
        cast_pad_bf16<<<ceil_div((size_t)TDIM * DVEC, T), T, 0, stream>>>(
            qkv_w + (size_t)l * TDIM * DVEC, TDIM, DVEC,
            wqkv_bf + (size_t)l * TDIM * DVEC, TDIM, DVEC);
        cast_pad_bf16<<<ceil_div((size_t)DVEC * DVEC, T), T, 0, stream>>>(
            proj_w + (size_t)l * DVEC * DVEC, DVEC, DVEC,
            wproj_bf + (size_t)l * DVEC * DVEC, DVEC, DVEC);
        cast_pad_bf16<<<ceil_div((size_t)MLPP * DVEC, T), T, 0, stream>>>(
            fc1_w + (size_t)l * MLPH * DVEC, MLPH, DVEC,
            wfc1_bf + (size_t)l * MLPP * DVEC, MLPP, DVEC);
        cast_pad_bf16<<<ceil_div((size_t)DVEC * MLPP, T), T, 0, stream>>>(
            fc2_w + (size_t)l * DVEC * MLPH, DVEC, MLPH,
            wfc2_bf + (size_t)l * DVEC * MLPP, DVEC, MLPP);
        pad_bias_kernel<<<ceil_div(MLPP, T), T, 0, stream>>>(
            fc1_b + (size_t)l * MLPH, fc1b_pad + (size_t)l * MLPP, MLPH, MLPP);
    }

    // ---- patch embed: im2col + GEMM + pos ----
    im2col_bf16<<<ceil_div((size_t)NTOK * KPATCH, T), T, 0, stream>>>(img, xp_bf);
    gemm_bf16_wmma<<<dim3(DVEC / 64, NTOK / 128), 128, 0, stream>>>(
        xp_bf, KPATCH, wp_bf, KPATCH, patch_b, nullptr, x, nullptr,
        DVEC, NTOK, DVEC, KPATCH, 0);
    add_pos_kernel<<<ceil_div((size_t)NTOK * DVEC, T), T, 0, stream>>>(x, pos);

    const float scale = 1.0f / sqrtf((float)DVEC);

    for (int l = 0; l < LBLK; ++l) {
        // LN1 -> bf16
        layernorm_kernel<<<NTOK, 256, 0, stream>>>(
            x, ln1_w + (size_t)l * DVEC, ln1_b + (size_t)l * DVEC, ln_bf, nullptr);
        // QKV GEMM: [N,D] x [3D,D]^T
        gemm_bf16_wmma<<<dim3(TDIM / 64, NTOK / 128), 128, 0, stream>>>(
            ln_bf, DVEC, wqkv_bf + (size_t)l * TDIM * DVEC, DVEC,
            qkv_b + (size_t)l * TDIM, nullptr, qkv, nullptr,
            TDIM, NTOK, TDIM, DVEC, 0);
        // split: Q*scale [N,D], K [N,D], V^T [D,N]
        extract_cast_kernel<<<ceil_div((size_t)NTOK * DVEC, T), T, 0, stream>>>(
            qkv, TDIM, 0, qb, NTOK, DVEC, scale, 0);
        extract_cast_kernel<<<ceil_div((size_t)NTOK * DVEC, T), T, 0, stream>>>(
            qkv, TDIM, DVEC, kb, NTOK, DVEC, 1.0f, 0);
        extract_cast_kernel<<<ceil_div((size_t)NTOK * DVEC, T), T, 0, stream>>>(
            qkv, TDIM, 2 * DVEC, vtb, NTOK, DVEC, 1.0f, 1);
        // S = (Q*scale) K^T : [N,D] x [N,D]^T
        gemm_bf16_wmma<<<dim3(NTOK / 64, NTOK / 128), 128, 0, stream>>>(
            qb, DVEC, kb, DVEC, nullptr, nullptr, S, nullptr,
            NTOK, NTOK, NTOK, DVEC, 0);
        // P = softmax(S), bf16
        softmax_kernel<<<NTOK, 256, 0, stream>>>(S, P);
        // O = P V : [N,N] x (V^T [D,N])^T  (bf16 out)
        gemm_bf16_wmma<<<dim3(DVEC / 64, NTOK / 128), 128, 0, stream>>>(
            P, NTOK, vtb, NTOK, nullptr, nullptr, nullptr, o_bf,
            DVEC, NTOK, DVEC, NTOK, 0);
        // x = x + O W_proj^T + b
        gemm_bf16_wmma<<<dim3(DVEC / 64, NTOK / 128), 128, 0, stream>>>(
            o_bf, DVEC, wproj_bf + (size_t)l * DVEC * DVEC, DVEC,
            proj_b + (size_t)l * DVEC, x, x, nullptr,
            DVEC, NTOK, DVEC, DVEC, 0);
        // LN2 -> bf16
        layernorm_kernel<<<NTOK, 256, 0, stream>>>(
            x, ln2_w + (size_t)l * DVEC, ln2_b + (size_t)l * DVEC, ln_bf, nullptr);
        // H = gelu(LN2 W_fc1^T + b)  (bf16 out; padded rows of W give exact 0)
        gemm_bf16_wmma<<<dim3(MLPP / 64, NTOK / 128), 128, 0, stream>>>(
            ln_bf, DVEC, wfc1_bf + (size_t)l * MLPP * DVEC, DVEC,
            fc1b_pad + (size_t)l * MLPP, nullptr, nullptr, h_bf,
            MLPP, NTOK, MLPP, DVEC, 1);
        // x = x + H W_fc2^T + b   (padded K cols of H are 0)
        gemm_bf16_wmma<<<dim3(DVEC / 64, NTOK / 128), 128, 0, stream>>>(
            h_bf, MLPP, wfc2_bf + (size_t)l * DVEC * MLPP, MLPP,
            fc2_b + (size_t)l * DVEC, x, x, nullptr,
            DVEC, NTOK, DVEC, MLPP, 0);
    }

    // final LN -> f32 output
    layernorm_kernel<<<NTOK, 256, 0, stream>>>(x, post_w, post_b, nullptr, out);
}
